// Seq2SeqPointAttn_29695403884755
// MI455X (gfx1250) — matrix operand order repaired and checked by reading, not compiled, not run on hardware
//
#include <hip/hip_runtime.h>
#include <hip/hip_bf16.h>
#include <math.h>

#define BB   64
#define LPRE 256
#define EMB  512
#define HID  1024
#define G3   3072
#define S2   512
#define MAXL 16

typedef _Float16 h8   __attribute__((ext_vector_type(8)));
typedef _Float16 v16h __attribute__((ext_vector_type(16)));
typedef float    v8f  __attribute__((ext_vector_type(8)));

// ---------------- WMMA helpers (CDNA5 16x16x32 f16 -> f32) ----------------

__device__ __forceinline__ v8f wmma32(v16h a, v16h b, v8f c) {
  // 8 args: (neg_a, A, neg_b, B, c_mod, C, reuse_a, reuse_b)
  return __builtin_amdgcn_wmma_f32_16x16x32_f16(false, a, false, b, (short)0, c,
                                                false, false);
}

// A operand, 16x32 f16 (ISA 7.12.2 "16-bit A-Matrix 16x32"):
// lane m (m<16): VGPR0-3 = K 0..7, VGPR4-7 = K 16..23
// lane m+16    : VGPR0-3 = K 8..15, VGPR4-7 = K 24..31
__device__ __forceinline__ v16h load_frag_a(const _Float16* __restrict__ p, int ld,
                                            int rowBase, int k0, int lane) {
  int r  = rowBase + (lane & 15);
  int kh = (lane >> 4) << 3;  // 0 or 8
  const _Float16* q = p + (size_t)r * (size_t)ld + (size_t)(k0 + kh);
  h8 lo = *(const h8*)q;        // K = k0+kh .. +7
  h8 hi = *(const h8*)(q + 16); // K = k0+kh+16 .. +23
  v16h f;
#pragma unroll
  for (int i = 0; i < 8; ++i) { f[i] = lo[i]; f[8 + i] = hi[i]; }
  return f;
}

// B operand, 32x16 f16 (ISA 7.12.4 B-matrix striping):
// lane n (n<16) holds column n, K = 0..7 then 16..23;
// lane n+16 holds column n, K = 8..15 then 24..31.
// Computing y = x*W^T: column n of B == row n of W (row-major (N,K)).
__device__ __forceinline__ v16h load_frag_b(const _Float16* __restrict__ p, int ld,
                                            int colBase, int k0, int lane) {
  int n  = colBase + (lane & 15);
  int kb = (lane >> 4) << 3;  // 0 or 8
  const _Float16* q = p + (size_t)n * (size_t)ld + (size_t)(k0 + kb);
  h8 lo = *(const h8*)q;       // K = k0+kb .. +7
  h8 hi = *(const h8*)(q + 16); // K = k0+kb+16 .. +23
  v16h f;
#pragma unroll
  for (int i = 0; i < 8; ++i) { f[i] = lo[i]; f[8 + i] = hi[i]; }
  return f;
}

// ---------------- Small utility kernels ----------------

__global__ void f32_to_f16_k(const float* __restrict__ s, _Float16* __restrict__ d) {
  int i = blockIdx.x * 256 + threadIdx.x;
  d[i] = (_Float16)s[i];
}

__global__ void zero_f32_k(float* __restrict__ d) {
  d[blockIdx.x * 256 + threadIdx.x] = 0.f;
}
__global__ void zero_f16_k(_Float16* __restrict__ d) {
  d[blockIdx.x * 256 + threadIdx.x] = (_Float16)0.f;
}

// dst[(b,t,e)] = (f16) emb[seq[b,t], e]
__global__ void gather_embed_k(const int* __restrict__ seq, const float* __restrict__ emb,
                               _Float16* __restrict__ dst) {
  int i  = blockIdx.x * 256 + threadIdx.x;
  int e  = i & (EMB - 1);
  int bt = i >> 9;
  dst[i] = (_Float16)emb[(size_t)seq[bt] * EMB + e];
}

// hcat[b, 0:1024] = h_pre[b], hcat[b, 1024:2048] = h_post[b]   (f16)
__global__ void hcat_pack_k(const float* __restrict__ hP, const float* __restrict__ hQ,
                            _Float16* __restrict__ hc) {
  int i = blockIdx.x * 256 + threadIdx.x;   // < 64*2048
  int b = i >> 11, j = i & 2047;
  float v = (j < HID) ? hP[b * HID + j] : hQ[b * HID + (j - HID)];
  hc[i] = (_Float16)v;
}

// ---------------- Generic WMMA GEMM:  out = act(A(MxK) * B(NxK)^T + bias + Cin) ----------------
// 8 waves / block, one 16x16 tile per wave. M,N % 16 == 0, K % 32 == 0.

__global__ void gemm_wmma_k(const _Float16* __restrict__ A, int lda,
                            const _Float16* __restrict__ Bw, int ldb,
                            const float* __restrict__ bias,
                            const float* __restrict__ Cin,
                            float* __restrict__ outF, _Float16* __restrict__ outH,
                            int M, int N, int K, int act) {
  int lane = threadIdx.x & 31, wid = threadIdx.x >> 5;
  int nT = N >> 4;
  int id = blockIdx.x * 8 + wid;
  int mt = id / nT, nt = id - mt * nT;
  int mBase = mt << 4, nBase = nt << 4;
  if (mBase >= M) return;                 // wave-uniform; EXEC stays all-ones
  v8f acc = {};
  for (int k0 = 0; k0 < K; k0 += 32) {
    v16h a = load_frag_a(A, lda, mBase, k0, lane);
    v16h b = load_frag_b(Bw, ldb, nBase, k0, lane);
    acc = wmma32(a, b, acc);
  }
  int c = nBase + (lane & 15);
  float bv = bias ? bias[c] : 0.f;
#pragma unroll
  for (int v = 0; v < 8; ++v) {
    int r = mBase + v + ((lane >> 4) << 3);
    float x = acc[v] + bv;
    if (Cin) x += Cin[(size_t)r * N + c];
    if (act == 1) x = tanhf(x);
    if (outF) outF[(size_t)r * N + c] = x;
    if (outH) outH[(size_t)r * N + c] = (_Float16)x;
  }
}

// ---------------- GRU step: gx = x_t*Wih^T, gh = h*Whh^T for both encoders ----------------
// grid (96,1,2), block 256.  Per block: 64 (M) x 32 (N) region, 8 waves = 4x2 tiles.

__global__ void gru_step_gemm_k(int t,
    const _Float16* __restrict__ xhP, const _Float16* __restrict__ xhQ,
    const _Float16* __restrict__ WihP, const _Float16* __restrict__ WihQ,
    const _Float16* __restrict__ WhhP, const _Float16* __restrict__ WhhQ,
    const _Float16* __restrict__ hhP,  const _Float16* __restrict__ hhQ,
    float* __restrict__ gxP, float* __restrict__ gxQ,
    float* __restrict__ ghP, float* __restrict__ ghQ) {
  int enc = blockIdx.z;
  const _Float16* xh  = enc ? xhQ  : xhP;
  const _Float16* Wih = enc ? WihQ : WihP;
  const _Float16* Whh = enc ? WhhQ : WhhP;
  const _Float16* hh  = enc ? hhQ  : hhP;
  float* gx = enc ? gxQ : gxP;
  float* gh = enc ? ghQ : ghP;

  int lane = threadIdx.x & 31, wid = threadIdx.x >> 5;
  int mBase = (wid & 3) << 4;
  int nBase = blockIdx.x * 32 + ((wid >> 2) << 4);

  const _Float16* A = xh + (size_t)t * EMB;  // row b at b*(T*E), lda = T*E
  v8f ax = {};
  for (int k0 = 0; k0 < EMB; k0 += 32)
    ax = wmma32(load_frag_a(A, LPRE * EMB, mBase, k0, lane),
                load_frag_b(Wih, EMB, nBase, k0, lane), ax);
  v8f ah = {};
  for (int k0 = 0; k0 < HID; k0 += 32)
    ah = wmma32(load_frag_a(hh, HID, mBase, k0, lane),
                load_frag_b(Whh, HID, nBase, k0, lane), ah);

  int c = nBase + (lane & 15);
#pragma unroll
  for (int v = 0; v < 8; ++v) {
    int r = mBase + v + ((lane >> 4) << 3);
    gx[r * G3 + c] = ax[v];
    gh[r * G3 + c] = ah[v];
  }
}

// Gate math (torch order r,z,n), h update, enc_out write. 131072 threads.
__global__ void gru_gates_k(int t,
    const float* __restrict__ gxP, const float* __restrict__ gxQ,
    const float* __restrict__ ghP, const float* __restrict__ ghQ,
    const float* __restrict__ bihP, const float* __restrict__ bhhP,
    const float* __restrict__ bihQ, const float* __restrict__ bhhQ,
    float* __restrict__ hP, float* __restrict__ hQ,
    _Float16* __restrict__ hhP, _Float16* __restrict__ hhQ,
    _Float16* __restrict__ enc_out) {
  int idx = blockIdx.x * 256 + threadIdx.x;
  int enc = idx >> 16;
  int rem = idx & 65535;
  int b = rem >> 10, j = rem & 1023;
  const float* gx  = enc ? gxQ  : gxP;
  const float* gh  = enc ? ghQ  : ghP;
  const float* bih = enc ? bihQ : bihP;
  const float* bhh = enc ? bhhQ : bhhP;
  float* h = enc ? hQ : hP;
  _Float16* hh = enc ? hhQ : hhP;
  int base = b * G3;
  float r = 1.f / (1.f + expf(-(gx[base + j] + bih[j] + gh[base + j] + bhh[j])));
  float z = 1.f / (1.f + expf(-(gx[base + HID + j] + bih[HID + j] +
                                gh[base + HID + j] + bhh[HID + j])));
  float n = tanhf(gx[base + 2 * HID + j] + bih[2 * HID + j] +
                  r * (gh[base + 2 * HID + j] + bhh[2 * HID + j]));
  float hprev = h[b * HID + j];
  float hn = (1.f - z) * n + z * hprev;
  h[b * HID + j]  = hn;
  hh[b * HID + j] = (_Float16)hn;
  int s = enc * LPRE + t;
  enc_out[((size_t)b * S2 + s) * HID + j] = (_Float16)hn;
}

// ---------------- Attention / head kernels ----------------

// scores[b,s] = sum_j tanh(hidA[b,j] + encA[b,s,j]) * v[j]   (one wave per (b,s))
__global__ void attn_scores_k(const float* __restrict__ hidA,
                              const _Float16* __restrict__ encA,
                              const float* __restrict__ vW,
                              float* __restrict__ scores) {
  int lane = threadIdx.x & 31, wid = threadIdx.x >> 5;
  int p = blockIdx.x * 8 + wid;  // < 32768
  int b = p >> 9;
  const _Float16* row = encA + (size_t)p * HID;
  const float* ha = hidA + (size_t)b * HID;
  float sum = 0.f;
  for (int j = lane; j < HID; j += 32)
    sum += tanhf(ha[j] + (float)row[j]) * vW[j];
#pragma unroll
  for (int m = 16; m; m >>= 1) sum += __shfl_xor(sum, m, 32);
  if (lane == 0) scores[p] = sum;
}

__global__ void softmax512_k(const float* __restrict__ sc, float* __restrict__ a) {
  __shared__ float red[256];
  int b = blockIdx.x, t = threadIdx.x;
  float x0 = sc[b * S2 + t], x1 = sc[b * S2 + t + 256];
  red[t] = fmaxf(x0, x1);
  __syncthreads();
  for (int s = 128; s; s >>= 1) { if (t < s) red[t] = fmaxf(red[t], red[t + s]); __syncthreads(); }
  float m = red[0];
  __syncthreads();
  float e0 = expf(x0 - m), e1 = expf(x1 - m);
  red[t] = e0 + e1;
  __syncthreads();
  for (int s = 128; s; s >>= 1) { if (t < s) red[t] += red[t + s]; __syncthreads(); }
  float inv = 1.f / red[0];
  a[b * S2 + t] = e0 * inv;
  a[b * S2 + t + 256] = e1 * inv;
}

// weighted[b,h] = sum_s a[b,s]*enc_out[b,s,h]
__global__ void weighted_k(const float* __restrict__ a, const _Float16* __restrict__ enc,
                           float* __restrict__ w, _Float16* __restrict__ wh) {
  __shared__ float as[S2];
  int b = blockIdx.x, t = threadIdx.x;
  as[t] = a[b * S2 + t];
  as[t + 256] = a[b * S2 + t + 256];
  __syncthreads();
#pragma unroll
  for (int k = 0; k < 4; ++k) {
    int hh = t + k * 256;
    const _Float16* base = enc + (size_t)b * S2 * HID + hh;
    float acc = 0.f;
    for (int s = 0; s < S2; ++s) acc += as[s] * (float)base[(size_t)s * HID];
    w[b * HID + hh]  = acc;
    wh[b * HID + hh] = (_Float16)acc;
  }
}

// cb[b] = fco_b + sum_j relu(o2[b,j]) * fcoW[HID+j]
__global__ void cb_k(const float* __restrict__ o2, const float* __restrict__ fcoW,
                     const float* __restrict__ fco_b, float* __restrict__ cb) {
  __shared__ float red[256];
  int b = blockIdx.x, t = threadIdx.x;
  float s = 0.f;
  for (int j = t; j < HID; j += 256) s += fmaxf(o2[b * HID + j], 0.f) * fcoW[HID + j];
  red[t] = s;
  __syncthreads();
  for (int k = 128; k; k >>= 1) { if (t < k) red[t] += red[t + k]; __syncthreads(); }
  if (t == 0) cb[b] = red[0] + fco_b[0];
}

// sig[b,s] broadcast over MAXL into outputs_pre / outputs_post
__global__ void logits_out_k(const _Float16* __restrict__ enc, const float* __restrict__ fcoW,
                             const float* __restrict__ cb, float* __restrict__ out) {
  int lane = threadIdx.x & 31, wid = threadIdx.x >> 5;
  int p = blockIdx.x * 8 + wid;   // (b,s), p < 32768
  int b = p >> 9, s = p & 511;
  const _Float16* row = enc + (size_t)p * HID;
  float sum = 0.f;
  for (int j = lane; j < HID; j += 32) sum += fmaxf((float)row[j], 0.f) * fcoW[j];
#pragma unroll
  for (int m = 16; m; m >>= 1) sum += __shfl_xor(sum, m, 32);
  if (lane == 0) {
    float sg = 1.f / (1.f + expf(-(sum + cb[b])));
    size_t base = (s < LPRE)
        ? ((size_t)b * MAXL * LPRE + s)
        : ((size_t)BB * MAXL * LPRE + (size_t)b * MAXL * LPRE + (s - LPRE));
#pragma unroll
    for (int tt = 0; tt < MAXL; ++tt) out[base + (size_t)tt * LPRE] = sg;
  }
}

// ---------------- Host orchestration ----------------

extern "C" void kernel_launch(void* const* d_in, const int* in_sizes, int n_in,
                              void* d_out, int out_size, void* d_ws, size_t ws_size,
                              hipStream_t stream) {
  (void)in_sizes; (void)n_in; (void)out_size; (void)ws_size;
  const int*   pre_seq  = (const int*)d_in[0];
  const int*   post_seq = (const int*)d_in[1];
  const float* emb      = (const float*)d_in[3];
  const float* Wih_pre  = (const float*)d_in[4];
  const float* Whh_pre  = (const float*)d_in[5];
  const float* bih_pre  = (const float*)d_in[6];
  const float* bhh_pre  = (const float*)d_in[7];
  const float* Wih_post = (const float*)d_in[8];
  const float* Whh_post = (const float*)d_in[9];
  const float* bih_post = (const float*)d_in[10];
  const float* bhh_post = (const float*)d_in[11];
  const float* fc_W   = (const float*)d_in[12];
  const float* fc_b   = (const float*)d_in[13];
  const float* attn_W = (const float*)d_in[14];
  const float* attn_b = (const float*)d_in[15];
  const float* v_W    = (const float*)d_in[16];
  const float* fc1_W  = (const float*)d_in[17];
  const float* fc1_b  = (const float*)d_in[18];
  const float* fc2_W  = (const float*)d_in[19];
  const float* fc2_b  = (const float*)d_in[20];
  const float* fco_W  = (const float*)d_in[21];
  const float* fco_b  = (const float*)d_in[22];
  float* out = (float*)d_out;

  // --- workspace carve-out (256B aligned slabs) ---
  char* w = (char*)d_ws;
  size_t off = 0;
  auto alloc = [&](size_t bytes) -> void* {
    off = (off + 255) & ~(size_t)255;
    void* p = w + off;
    off += bytes;
    return p;
  };
  _Float16* xh_pre  = (_Float16*)alloc((size_t)BB * LPRE * EMB * 2);
  _Float16* xh_post = (_Float16*)alloc((size_t)BB * LPRE * EMB * 2);
  _Float16* WihP_h  = (_Float16*)alloc((size_t)G3 * EMB * 2);
  _Float16* WhhP_h  = (_Float16*)alloc((size_t)G3 * HID * 2);
  _Float16* WihQ_h  = (_Float16*)alloc((size_t)G3 * EMB * 2);
  _Float16* WhhQ_h  = (_Float16*)alloc((size_t)G3 * HID * 2);
  _Float16* fcW_h   = (_Float16*)alloc((size_t)HID * 2 * HID * 2);
  _Float16* attnW_h = (_Float16*)alloc((size_t)HID * 2 * HID * 2);
  _Float16* fc1W_h  = (_Float16*)alloc((size_t)HID * HID * 2);
  _Float16* fc2W_h  = (_Float16*)alloc((size_t)HID * 2 * HID * 2);
  float*    hP      = (float*)alloc((size_t)BB * HID * 4);
  float*    hQ      = (float*)alloc((size_t)BB * HID * 4);
  _Float16* hhP     = (_Float16*)alloc((size_t)BB * HID * 2);
  _Float16* hhQ     = (_Float16*)alloc((size_t)BB * HID * 2);
  float*    gxP     = (float*)alloc((size_t)BB * G3 * 4);
  float*    gxQ     = (float*)alloc((size_t)BB * G3 * 4);
  float*    ghP     = (float*)alloc((size_t)BB * G3 * 4);
  float*    ghQ     = (float*)alloc((size_t)BB * G3 * 4);
  _Float16* enc_h   = (_Float16*)alloc((size_t)BB * S2 * HID * 2);
  _Float16* hcat_h  = (_Float16*)alloc((size_t)BB * 2 * HID * 2);
  float*    hidden  = (float*)alloc((size_t)BB * HID * 4);
  _Float16* hidden_h= (_Float16*)alloc((size_t)BB * HID * 2);
  float*    hidA    = (float*)alloc((size_t)BB * HID * 4);
  _Float16* encA_h  = (_Float16*)alloc((size_t)BB * S2 * HID * 2);
  float*    scores  = (float*)alloc((size_t)BB * S2 * 4);
  float*    attn_a  = (float*)alloc((size_t)BB * S2 * 4);
  float*    wsum    = (float*)alloc((size_t)BB * HID * 4);
  _Float16* wsum_h  = (_Float16*)alloc((size_t)BB * HID * 2);
  _Float16* o_h     = (_Float16*)alloc((size_t)BB * HID * 2);
  float*    tmp2    = (float*)alloc((size_t)BB * HID * 4);
  float*    o2      = (float*)alloc((size_t)BB * HID * 4);
  float*    cb      = (float*)alloc(256);

  // --- embeddings -> f16 activations ---
  gather_embed_k<<<(BB * LPRE * EMB) / 256, 256, 0, stream>>>(pre_seq, emb, xh_pre);
  gather_embed_k<<<(BB * LPRE * EMB) / 256, 256, 0, stream>>>(post_seq, emb, xh_post);

  // --- weights -> f16 ---
  auto conv = [&](const float* s, _Float16* d, size_t n) {
    f32_to_f16_k<<<(unsigned)(n / 256), 256, 0, stream>>>(s, d);
  };
  conv(Wih_pre,  WihP_h,  (size_t)G3 * EMB);
  conv(Whh_pre,  WhhP_h,  (size_t)G3 * HID);
  conv(Wih_post, WihQ_h,  (size_t)G3 * EMB);
  conv(Whh_post, WhhQ_h,  (size_t)G3 * HID);
  conv(fc_W,   fcW_h,   (size_t)HID * 2 * HID);
  conv(attn_W, attnW_h, (size_t)HID * 2 * HID);
  conv(fc1_W,  fc1W_h,  (size_t)HID * HID);
  conv(fc2_W,  fc2W_h,  (size_t)HID * 2 * HID);

  // --- h0 = 0 (must re-init every call) ---
  zero_f32_k<<<(BB * HID) / 256, 256, 0, stream>>>(hP);
  zero_f32_k<<<(BB * HID) / 256, 256, 0, stream>>>(hQ);
  zero_f16_k<<<(BB * HID) / 256, 256, 0, stream>>>(hhP);
  zero_f16_k<<<(BB * HID) / 256, 256, 0, stream>>>(hhQ);

  // --- GRU scan: both encoders fused per step ---
  for (int t = 0; t < LPRE; ++t) {
    gru_step_gemm_k<<<dim3(96, 1, 2), 256, 0, stream>>>(
        t, xh_pre, xh_post, WihP_h, WihQ_h, WhhP_h, WhhQ_h, hhP, hhQ,
        gxP, gxQ, ghP, ghQ);
    gru_gates_k<<<512, 256, 0, stream>>>(
        t, gxP, gxQ, ghP, ghQ, bih_pre, bhh_pre, bih_post, bhh_post,
        hP, hQ, hhP, hhQ, enc_h);
  }

  // --- hidden = tanh([h_pre, h_post] @ fc_W^T + fc_b) ---
  hcat_pack_k<<<(BB * 2 * HID) / 256, 256, 0, stream>>>(hP, hQ, hcat_h);
  gemm_wmma_k<<<(4 * 64) / 8, 256, 0, stream>>>(hcat_h, 2 * HID, fcW_h, 2 * HID,
      fc_b, nullptr, hidden, hidden_h, BB, HID, 2 * HID, 1);

  // --- attention: hidA = hidden @ W1^T ; encA = enc_out @ W2^T + attn_b ---
  gemm_wmma_k<<<(4 * 64) / 8, 256, 0, stream>>>(hidden_h, HID, attnW_h, 2 * HID,
      nullptr, nullptr, hidA, nullptr, BB, HID, HID, 0);
  gemm_wmma_k<<<(2048 * 64) / 8, 256, 0, stream>>>(enc_h, HID, attnW_h + HID, 2 * HID,
      attn_b, nullptr, nullptr, encA_h, BB * S2, HID, HID, 0);
  attn_scores_k<<<(BB * S2) / 8, 256, 0, stream>>>(hidA, encA_h, v_W, scores);
  softmax512_k<<<BB, 256, 0, stream>>>(scores, attn_a);
  weighted_k<<<BB, 256, 0, stream>>>(attn_a, enc_h, wsum, wsum_h);

  // --- o = tanh(hidden@fc1^T+b1); o2 = tanh([o,weighted]@fc2^T+b2) ---
  gemm_wmma_k<<<(4 * 64) / 8, 256, 0, stream>>>(hidden_h, HID, fc1W_h, HID,
      fc1_b, nullptr, nullptr, o_h, BB, HID, HID, 1);
  gemm_wmma_k<<<(4 * 64) / 8, 256, 0, stream>>>(o_h, HID, fc2W_h, 2 * HID,
      fc2_b, nullptr, tmp2, nullptr, BB, HID, HID, 0);
  gemm_wmma_k<<<(4 * 64) / 8, 256, 0, stream>>>(wsum_h, HID, fc2W_h + HID, 2 * HID,
      nullptr, tmp2, o2, nullptr, BB, HID, HID, 1);

  // --- logits -> sigmoid -> broadcast outputs ---
  cb_k<<<BB, 256, 0, stream>>>(o2, fco_W, fco_b, cb);
  logits_out_k<<<(BB * S2) / 8, 256, 0, stream>>>(enc_h, fco_W, cb, out);
}
